// DGCNN_50130858279643
// MI455X (gfx1250) — compile-verified, matrix operand order checked by reading, not run on hardware
//
#include <hip/hip_runtime.h>
#include <hip/hip_bf16.h>

// ---------------- problem constants ----------------
#define BB 16
#define NN 1024
#define KN 20          // neighbors
#define EPSB 1e-5f
#define NEG_INF -3.0e38f
#define POS_INF  3.0e38f

typedef __attribute__((ext_vector_type(16))) _Float16 v16h;
typedef __attribute__((ext_vector_type(8)))  _Float16 v8h;
typedef __attribute__((ext_vector_type(8)))  float    v8f;

// Load a 16x32 f16 fragment (A or B operand) from a row-major [rows, ld] f16
// matrix: 16 rows starting at r0, K-chunk starting at kk (ld, kk multiples of 32,
// base 16B aligned).  Matches CDNA5 16-bit operand layout:
//   lanes 0-15 : row = lane,    halves 0..7 = K kk..kk+7,  8..15 = K kk+16..kk+23
//   lanes16-31 : row = lane-16, halves 0..7 = K kk+8..+15, 8..15 = K kk+24..+31
__device__ __forceinline__ v16h load_frag(const _Float16* __restrict__ m,
                                          int ld, int r0, int kk) {
    int lane = threadIdx.x & 31;
    int sel  = lane >> 4;
    int r    = lane & 15;
    const _Float16* p = m + (size_t)(r0 + r) * ld + kk + sel * 8;
    v8h lo = *(const v8h*)(p);
    v8h hi = *(const v8h*)(p + 16);
    v16h out;
#pragma unroll
    for (int i = 0; i < 8; ++i) { out[i] = lo[i]; out[8 + i] = hi[i]; }
    return out;
}

// One 8-half run of the edge feature e = [h(nbr)-h(ctr) ; h(ctr) ; 0pad] at
// channel offset k0 (multiple of 8).  For ci in {64,128} every run is entirely
// inside one region -> two 16B loads + packed f16 subtract; layer 1 (ci=3)
// straddles only in run k0=0 -> scalar fallback.
__device__ __forceinline__ v8h edge_run(const _Float16* __restrict__ hc,
                                        const _Float16* __restrict__ hn,
                                        int k0, int ci, int ci2) {
    if (k0 + 8 <= ci) {
        return *(const v8h*)(hn + k0) - *(const v8h*)(hc + k0);
    }
    if (k0 >= ci2) {
        v8h z = {};
        return z;
    }
    if (k0 >= ci && k0 + 8 <= ci2) {           // ci multiple of 8 here -> aligned
        return *(const v8h*)(hc + (k0 - ci));
    }
    v8h r;                                     // straddle (layer 1 only)
#pragma unroll
    for (int t = 0; t < 8; ++t) {
        int kg = k0 + t;
        r[t] = (kg < ci) ? (_Float16)(hn[kg] - hc[kg])
             : ((kg < ci2) ? hc[kg - ci] : (_Float16)0.f);
    }
    return r;
}

// ---------------- stage 0: input prep ----------------
// x [B,N,3] f32 -> hT [B*N, 32] f16 (zero padded), xx [B*N] = |x|^2
__global__ void k_prep(const float* __restrict__ x, _Float16* __restrict__ hT,
                       float* __restrict__ xx) {
    unsigned t = blockIdx.x * blockDim.x + threadIdx.x;
    if (t >= BB * NN) return;
    const float* xp = x + (size_t)t * 3;
    _Float16* hp = hT + (size_t)t * 32;
    float s = 0.f;
#pragma unroll
    for (int c = 0; c < 32; ++c) {
        float v = (c < 3) ? xp[c] : 0.f;
        hp[c] = (_Float16)v;
        if (c < 3) s += v * v;
    }
    xx[t] = s;
}

// hT [B*N, Cpad] f16 -> xx [B*N]
__global__ void k_sqnorm(const _Float16* __restrict__ hT, int Cpad,
                         float* __restrict__ xx) {
    unsigned t = blockIdx.x * blockDim.x + threadIdx.x;
    if (t >= BB * NN) return;
    const _Float16* p = hT + (size_t)t * Cpad;
    float s = 0.f;
    for (int c = 0; c < Cpad; ++c) { float v = (float)p[c]; s += v * v; }
    xx[t] = s;
}

// ---------------- Gram / negative squared distance (WMMA) ----------------
// dist[b,i,j] = 2 * h_i . h_j - |h_i|^2 - |h_j|^2   (one wave per 16x16 tile)
__global__ void k_gram(const _Float16* __restrict__ hT, const float* __restrict__ xx,
                       float* __restrict__ dist, int Cpad) {
    int tj = blockIdx.x, ti = blockIdx.y, b = blockIdx.z;
    const _Float16* hb = hT + (size_t)b * NN * Cpad;
    int row0 = ti * 16, col0 = tj * 16;
    v8f acc = {};
    for (int kk = 0; kk < Cpad; kk += 32) {
        v16h a  = load_frag(hb, Cpad, row0, kk);
        v16h bf = load_frag(hb, Cpad, col0, kk);
        acc = __builtin_amdgcn_wmma_f32_16x16x32_f16(false, a, false, bf,
                                                     (short)0, acc, false, false);
    }
    int lane = threadIdx.x & 31;
    int sel = lane >> 4, nc = lane & 15;
    const float* xb = xx + b * NN;
    float xj = xb[col0 + nc];
    float* db = dist + ((size_t)b * NN + row0) * NN;
#pragma unroll
    for (int r = 0; r < 8; ++r) {
        int M = r + sel * 8;
        db[(size_t)M * NN + col0 + nc] = 2.f * acc[r] - xb[row0 + M] - xj;
    }
}

// ---------------- top-k (k=20) per point ----------------
__global__ void k_topk(const float* __restrict__ dist, int* __restrict__ idx) {
    unsigned t = blockIdx.x * blockDim.x + threadIdx.x;
    if (t >= BB * NN) return;
    const float* row = dist + (size_t)t * NN;
    float vals[KN];
    int ids[KN];
#pragma unroll
    for (int i = 0; i < KN; ++i) { vals[i] = NEG_INF; ids[i] = 0; }
    for (int j = 0; j < NN; ++j) {
        float v = row[j];
        if (v > vals[KN - 1]) {
            int p = KN - 1;
            while (p > 0 && vals[p - 1] < v) {
                vals[p] = vals[p - 1]; ids[p] = ids[p - 1]; --p;
            }
            vals[p] = v; ids[p] = j;
        }
    }
    int* o = idx + (size_t)t * KN;
#pragma unroll
    for (int i = 0; i < KN; ++i) o[i] = ids[i];
}

// ---------------- weight convert f32 [cout, kin] -> f16 [cout, Kpad] ----------------
__global__ void k_cvtW(const float* __restrict__ w, _Float16* __restrict__ wp,
                       int kin, int Kpad, unsigned total) {
    unsigned t = blockIdx.x * blockDim.x + threadIdx.x;
    if (t >= total) return;
    int k = t % Kpad, o = t / Kpad;
    wp[t] = (_Float16)((k < kin) ? w[(size_t)o * kin + k] : 0.f);
}

// ---------------- edge conv (WMMA), one wave per (cout-tile, point) ----------------
// Neighbor dim padded 20->32: two 16-wide B tiles / two accumulators per wave.
// Wave owns all k for (b, n, 16 channels): max/min over k are in-wave shfl
// reductions + plain stores (no atomics, no init).  Invalid kb lanes use self
// as dummy neighbor and are masked out of max/min/sum.
__global__ void k_edgeconv(const _Float16* __restrict__ Wp,
                           const _Float16* __restrict__ hT, int Cpad, int ci,
                           const int* __restrict__ idx,
                           float* __restrict__ sum, float* __restrict__ sumsq,
                           float* __restrict__ mx, float* __restrict__ mn,
                           int cout, int Kpad) {
    int mt = blockIdx.x;           // cout / 16
    int n  = blockIdx.y;           // point
    int b  = blockIdx.z;
    int lane = threadIdx.x & 31;
    int sel = lane >> 4, l = lane & 15;
    const int* ip = idx + ((size_t)b * NN + n) * KN;
    int nb0 = ip[l];                               // kb = l      (always valid)
    bool v1 = (l < KN - 16);                       // kb = 16 + l (valid l<4)
    int nb1 = v1 ? ip[16 + l] : n;
    const _Float16* hc  = hT + ((size_t)b * NN + n)   * Cpad;
    const _Float16* hn0 = hT + ((size_t)b * NN + nb0) * Cpad;
    const _Float16* hn1 = hT + ((size_t)b * NN + nb1) * Cpad;
    int ci2 = 2 * ci;

    v8f acc0 = {}, acc1 = {};
    for (int kk = 0; kk < Kpad; kk += 32) {
        v16h a = load_frag(Wp, Kpad, mt * 16, kk);
        int kb8 = kk + sel * 8;
        v8h lo0 = edge_run(hc, hn0, kb8,      ci, ci2);
        v8h hi0 = edge_run(hc, hn0, kb8 + 16, ci, ci2);
        v8h lo1 = edge_run(hc, hn1, kb8,      ci, ci2);
        v8h hi1 = edge_run(hc, hn1, kb8 + 16, ci, ci2);
        v16h b0, b1;
#pragma unroll
        for (int i = 0; i < 8; ++i) {
            b0[i] = lo0[i]; b0[8 + i] = hi0[i];
            b1[i] = lo1[i]; b1[8 + i] = hi1[i];
        }
        acc0 = __builtin_amdgcn_wmma_f32_16x16x32_f16(false, a, false, b0,
                                                      (short)0, acc0, false, false);
        acc1 = __builtin_amdgcn_wmma_f32_16x16x32_f16(false, a, false, b1,
                                                      (short)0, acc1, false, false);
    }
#pragma unroll
    for (int r = 0; r < 8; ++r) {
        int c = mt * 16 + r + sel * 8;
        float a0 = acc0[r], a1 = acc1[r];
        float cmx = v1 ? fmaxf(a0, a1) : a0;
        float cmn = v1 ? fminf(a0, a1) : a0;
        float s   = a0 + (v1 ? a1 : 0.f);
        float s2  = a0 * a0 + (v1 ? a1 * a1 : 0.f);
#pragma unroll
        for (int m = 1; m < 16; m <<= 1) {
            cmx = fmaxf(cmx, __shfl_xor(cmx, m, 32));
            cmn = fminf(cmn, __shfl_xor(cmn, m, 32));
            s  += __shfl_xor(s,  m, 32);
            s2 += __shfl_xor(s2, m, 32);
        }
        if (l == 0) {
            size_t mi = ((size_t)b * cout + c) * NN + n;
            mx[mi] = cmx;
            mn[mi] = cmn;
            atomicAdd(&sum[c], s);
            atomicAdd(&sumsq[c], s2);
        }
    }
}

// ---------------- BN scale/bias from raw stats ----------------
__global__ void k_make_sb(const float* __restrict__ sum, const float* __restrict__ sumsq,
                          const float* __restrict__ gamma, const float* __restrict__ beta,
                          float invcount, int C,
                          float* __restrict__ scale, float* __restrict__ bias) {
    int c = blockIdx.x * blockDim.x + threadIdx.x;
    if (c >= C) return;
    float m   = sum[c] * invcount;
    float var = sumsq[c] * invcount - m * m;
    float a = gamma[c] * rsqrtf(var + EPSB);
    scale[c] = a;
    bias[c]  = beta[c] - a * m;
}

// ---------------- apply BN+leaky to k-extremum; emit next-layer f16 features ----------------
// (max over k commutes with monotone BN∘leaky when scale>0; use min when scale<0)
__global__ void k_edge_apply(const float* __restrict__ mx, const float* __restrict__ mn,
                             const float* __restrict__ scale, const float* __restrict__ bias,
                             _Float16* __restrict__ hTnext, int CpadNext,
                             _Float16* __restrict__ catT, int catOff, int cout) {
    unsigned t = blockIdx.x * blockDim.x + threadIdx.x;
    if (t >= (unsigned)(BB * NN) * cout) return;
    int n = t % NN;
    int c = (t / NN) % cout;
    int b = t / (NN * cout);
    float a = scale[c];
    float v = (a >= 0.f) ? mx[t] : mn[t];
    float y = a * v + bias[c];
    y = (y >= 0.f) ? y : 0.2f * y;
    _Float16 hv = (_Float16)y;
    hTnext[((size_t)b * NN + n) * CpadNext + c] = hv;
    catT[((size_t)b * NN + n) * 512 + catOff + c] = hv;
}

// ---------------- aggregation conv 512->1024 (WMMA) with fused stats ----------------
__global__ void k_aggconv(const _Float16* __restrict__ Wp,   // [1024,512] f16
                          const _Float16* __restrict__ catT, // [B*N,512] f16
                          float* __restrict__ z,             // [B,1024,N] f32
                          float* __restrict__ sum, float* __restrict__ sumsq) {
    int mt = blockIdx.x;   // 64
    int ct = blockIdx.y;   // 1024 column tiles (B*N/16)
    int lane = threadIdx.x & 31;
    int sel = lane >> 4, l = lane & 15;
    v8f acc = {};
    for (int kk = 0; kk < 512; kk += 32) {
        v16h a  = load_frag(Wp, 512, mt * 16, kk);
        v16h bf = load_frag(catT, 512, ct * 16, kk);
        acc = __builtin_amdgcn_wmma_f32_16x16x32_f16(false, a, false, bf,
                                                     (short)0, acc, false, false);
    }
    int col = ct * 16 + l;
    int b = col >> 10, n = col & (NN - 1);
#pragma unroll
    for (int r = 0; r < 8; ++r) {
        int c = mt * 16 + r + sel * 8;
        float v = acc[r];
        z[((size_t)b * 1024 + c) * NN + n] = v;
        float s = v, s2 = v * v;
#pragma unroll
        for (int m = 1; m < 16; m <<= 1) {
            s  += __shfl_xor(s,  m, 32);
            s2 += __shfl_xor(s2, m, 32);
        }
        if (l == 0) { atomicAdd(&sum[c], s); atomicAdd(&sumsq[c], s2); }
    }
}

// ---------------- BN+leaky then max & mean pool over N ----------------
__global__ void k_agg_pool(const float* __restrict__ z,
                           const float* __restrict__ scale, const float* __restrict__ bias,
                           float* __restrict__ pooled) {
    int c = blockIdx.x, b = blockIdx.y;
    const float* zp = z + ((size_t)b * 1024 + c) * NN;
    float a = scale[c], bs = bias[c];
    float mxv = NEG_INF, sm = 0.f;
    for (int n = threadIdx.x; n < NN; n += blockDim.x) {
        float y = a * zp[n] + bs;
        y = (y >= 0.f) ? y : 0.2f * y;
        mxv = fmaxf(mxv, y);
        sm += y;
    }
    __shared__ float smx[256], ssm[256];
    int tid = threadIdx.x;
    smx[tid] = mxv; ssm[tid] = sm;
    __syncthreads();
    for (int s = 128; s > 0; s >>= 1) {
        if (tid < s) { smx[tid] = fmaxf(smx[tid], smx[tid + s]); ssm[tid] += ssm[tid + s]; }
        __syncthreads();
    }
    if (tid == 0) {
        pooled[(size_t)b * 2048 + c]        = smx[0];
        pooled[(size_t)b * 2048 + 1024 + c] = ssm[0] * (1.f / NN);
    }
}

// ---------------- head FCs (small) ----------------
__global__ void k_fc_stats(const float* __restrict__ in, int din,
                           const float* __restrict__ w, const float* __restrict__ bias,
                           int dout, float* __restrict__ pre,
                           float* __restrict__ sum, float* __restrict__ sumsq) {
    unsigned t = blockIdx.x * blockDim.x + threadIdx.x;
    if (t >= (unsigned)(BB * dout)) return;
    int o = t % dout, b = t / dout;
    const float* ip = in + (size_t)b * din;
    const float* wp = w + (size_t)o * din;
    float s = bias[o];
    for (int i = 0; i < din; ++i) s += ip[i] * wp[i];
    pre[t] = s;
    atomicAdd(&sum[o], s);
    atomicAdd(&sumsq[o], s * s);
}

__global__ void k_bn_leaky(const float* __restrict__ pre,
                           const float* __restrict__ scale, const float* __restrict__ bias,
                           float* __restrict__ out, int dout) {
    unsigned t = blockIdx.x * blockDim.x + threadIdx.x;
    if (t >= (unsigned)(BB * dout)) return;
    int o = t % dout;
    float y = scale[o] * pre[t] + bias[o];
    out[t] = (y >= 0.f) ? y : 0.2f * y;
}

__global__ void k_fc_final(const float* __restrict__ in, const float* __restrict__ w,
                           const float* __restrict__ bias, float* __restrict__ out) {
    unsigned t = blockIdx.x * blockDim.x + threadIdx.x;
    if (t >= BB * 40) return;
    int o = t % 40, b = t / 40;
    const float* ip = in + (size_t)b * 256;
    const float* wp = w + (size_t)o * 256;
    float s = bias[o];
    for (int i = 0; i < 256; ++i) s += ip[i] * wp[i];
    out[t] = s;
}

// ---------------- host side ----------------
static inline void* wsp(void* base, size_t& off, size_t bytes) {
    void* p = (char*)base + off;
    off += (bytes + 255) & ~(size_t)255;
    return p;
}

extern "C" void kernel_launch(void* const* d_in, const int* in_sizes, int n_in,
                              void* d_out, int out_size, void* d_ws, size_t ws_size,
                              hipStream_t stream) {
    (void)in_sizes; (void)n_in; (void)out_size; (void)ws_size;
    // inputs in setup_inputs() dict order (lists flattened):
    const float* x = (const float*)d_in[0];
    const float* conv_w[4] = {(const float*)d_in[1], (const float*)d_in[2],
                              (const float*)d_in[3], (const float*)d_in[4]};
    const float* conv_g[4] = {(const float*)d_in[5], (const float*)d_in[6],
                              (const float*)d_in[7], (const float*)d_in[8]};
    const float* conv_b[4] = {(const float*)d_in[9], (const float*)d_in[10],
                              (const float*)d_in[11], (const float*)d_in[12]};
    const float* agg_w = (const float*)d_in[13];
    const float* agg_g = (const float*)d_in[14];
    const float* agg_b = (const float*)d_in[15];
    const float* hw[3] = {(const float*)d_in[16], (const float*)d_in[17], (const float*)d_in[18]};
    const float* hb[3] = {(const float*)d_in[19], (const float*)d_in[20], (const float*)d_in[21]};
    const float* hg[2] = {(const float*)d_in[22], (const float*)d_in[23]};
    const float* hbt[2] = {(const float*)d_in[24], (const float*)d_in[25]};

    // workspace carve-out
    size_t off = 0;
    float*     dist  = (float*)    wsp(d_ws, off, (size_t)BB * NN * NN * 4); // reused as z
    int*       idx   = (int*)      wsp(d_ws, off, (size_t)BB * NN * KN * 4);
    _Float16*  hA    = (_Float16*) wsp(d_ws, off, (size_t)BB * NN * 256 * 2);
    _Float16*  hBuf  = (_Float16*) wsp(d_ws, off, (size_t)BB * NN * 256 * 2);
    _Float16*  catT  = (_Float16*) wsp(d_ws, off, (size_t)BB * NN * 512 * 2);
    float*     xx    = (float*)    wsp(d_ws, off, (size_t)BB * NN * 4);
    _Float16*  Wp    = (_Float16*) wsp(d_ws, off, (size_t)1024 * 512 * 2);
    float*     sum   = (float*)    wsp(d_ws, off, 1024 * 4);
    float*     sumsq = (float*)    wsp(d_ws, off, 1024 * 4);
    float*     scale = (float*)    wsp(d_ws, off, 1024 * 4);
    float*     bias  = (float*)    wsp(d_ws, off, 1024 * 4);
    float*     mx    = (float*)    wsp(d_ws, off, (size_t)BB * 256 * NN * 4);
    float*     mn    = (float*)    wsp(d_ws, off, (size_t)BB * 256 * NN * 4);
    float*     pool  = (float*)    wsp(d_ws, off, (size_t)BB * 2048 * 4);
    float*     pre1  = (float*)    wsp(d_ws, off, (size_t)BB * 512 * 4);
    float*     zh1   = (float*)    wsp(d_ws, off, (size_t)BB * 512 * 4);
    float*     pre2  = (float*)    wsp(d_ws, off, (size_t)BB * 256 * 4);
    float*     zh2   = (float*)    wsp(d_ws, off, (size_t)BB * 256 * 4);

    const int ci[4]   = {3, 64, 64, 128};
    const int cpin[4] = {32, 64, 64, 128};
    const int co[4]   = {64, 64, 128, 256};
    const int kpad[4] = {32, 128, 128, 256};
    const int coff[4] = {0, 64, 128, 256};

    _Float16* hbuf[2] = {hA, hBuf};
    int cur = 0;

    const unsigned PT = BB * NN; // points
    k_prep<<<(PT + 255) / 256, 256, 0, stream>>>(x, hbuf[0], xx);

    for (int L = 0; L < 4; ++L) {
        // kNN via WMMA Gram
        k_gram<<<dim3(64, 64, BB), 32, 0, stream>>>(hbuf[cur], xx, dist, cpin[L]);
        k_topk<<<(PT + 127) / 128, 128, 0, stream>>>(dist, idx);
        // weights -> f16 padded
        unsigned wtot = (unsigned)co[L] * kpad[L];
        k_cvtW<<<(wtot + 255) / 256, 256, 0, stream>>>(conv_w[L], Wp, 2 * ci[L], kpad[L], wtot);
        // reset per-channel stats
        hipMemsetAsync(sum,   0, co[L] * 4, stream);
        hipMemsetAsync(sumsq, 0, co[L] * 4, stream);
        // fused edge-conv GEMM: one wave per (cout-tile, point)
        k_edgeconv<<<dim3(co[L] / 16, NN, BB), 32, 0, stream>>>(
            Wp, hbuf[cur], cpin[L], ci[L], idx, sum, sumsq, mx, mn, co[L], kpad[L]);
        // BN affine
        float invc = 1.f / ((float)BB * NN * KN);
        k_make_sb<<<(co[L] + 127) / 128, 128, 0, stream>>>(sum, sumsq, conv_g[L], conv_b[L],
                                                           invc, co[L], scale, bias);
        // apply + produce next-layer features (f16) and concat slice
        unsigned etot = (unsigned)BB * co[L] * NN;
        k_edge_apply<<<(etot + 255) / 256, 256, 0, stream>>>(mx, mn, scale, bias,
                                                             hbuf[1 - cur], co[L],
                                                             catT, coff[L], co[L]);
        cur ^= 1;
        if (L < 3)
            k_sqnorm<<<(PT + 255) / 256, 256, 0, stream>>>(hbuf[cur], co[L], xx);
    }

    // aggregation conv 512 -> 1024 (reuse dist buffer as z)
    float* z = dist;
    unsigned awtot = 1024u * 512u;
    k_cvtW<<<(awtot + 255) / 256, 256, 0, stream>>>(agg_w, Wp, 512, 512, awtot);
    hipMemsetAsync(sum,   0, 1024 * 4, stream);
    hipMemsetAsync(sumsq, 0, 1024 * 4, stream);
    k_aggconv<<<dim3(64, (BB * NN) / 16), 32, 0, stream>>>(Wp, catT, z, sum, sumsq);
    k_make_sb<<<(1024 + 127) / 128, 128, 0, stream>>>(sum, sumsq, agg_g, agg_b,
                                                      1.f / ((float)BB * NN), 1024, scale, bias);
    k_agg_pool<<<dim3(1024, BB), 256, 0, stream>>>(z, scale, bias, pool);

    // head MLP
    hipMemsetAsync(sum,   0, 512 * 4, stream);
    hipMemsetAsync(sumsq, 0, 512 * 4, stream);
    k_fc_stats<<<(BB * 512 + 255) / 256, 256, 0, stream>>>(pool, 2048, hw[0], hb[0], 512,
                                                           pre1, sum, sumsq);
    k_make_sb<<<(512 + 127) / 128, 128, 0, stream>>>(sum, sumsq, hg[0], hbt[0],
                                                     1.f / (float)BB, 512, scale, bias);
    k_bn_leaky<<<(BB * 512 + 255) / 256, 256, 0, stream>>>(pre1, scale, bias, zh1, 512);

    hipMemsetAsync(sum,   0, 256 * 4, stream);
    hipMemsetAsync(sumsq, 0, 256 * 4, stream);
    k_fc_stats<<<(BB * 256 + 255) / 256, 256, 0, stream>>>(zh1, 512, hw[1], hb[1], 256,
                                                           pre2, sum, sumsq);
    k_make_sb<<<(256 + 127) / 128, 128, 0, stream>>>(sum, sumsq, hg[1], hbt[1],
                                                     1.f / (float)BB, 256, scale, bias);
    k_bn_leaky<<<(BB * 256 + 255) / 256, 256, 0, stream>>>(pre2, scale, bias, zh2, 256);

    k_fc_final<<<(BB * 40 + 63) / 64, 64, 0, stream>>>(zh2, hw[2], hb[2], (float*)d_out);
}
